// PuzzleSim_69097433858194
// MI455X (gfx1250) — matrix operand order repaired and checked by reading, not compiled, not run on hardware
//
#include <hip/hip_runtime.h>
#include <hip/hip_bf16.h>

typedef __attribute__((ext_vector_type(16))) __bf16 v16bf;
typedef __attribute__((ext_vector_type(8)))  float  v8f;

namespace {
constexpr int kN  = 32;
constexpr int kC  = 256;
constexpr int kH  = 64;
constexpr int kW  = 64;
constexpr int kHW = kH * kW;        // 4096 query pixels
constexpr int kR  = kN * kHW;       // 131072 reference rows
constexpr float kEps = 1e-12f;

constexpr int kQB = 128;            // queries per block (4 q-pairs x 32)
constexpr int kRB = 2048;           // reference rows per block slab
constexpr int kKC = kC / 32;        // 8 k-chunks of 32 (bf16 WMMA K)
}

union FragU {
  uint4 u[2];
  v16bf v;
};

// ---- normalize img over C, write bf16 queries Q[pixel][c] row-major ----
__global__ void norm_img_kernel(const float* __restrict__ img,
                                __bf16* __restrict__ Q) {
  int t = blockIdx.x * blockDim.x + threadIdx.x;   // pixel 0..4095
  float ss = 0.f;
  for (int c = 0; c < kC; ++c) {
    float x = img[(size_t)c * kHW + t];
    ss += x * x;
  }
  float inv = 1.f / fmaxf(sqrtf(ss), kEps);
  for (int c = 0; c < kC; ++c)
    Q[(size_t)t * kC + c] = (__bf16)(img[(size_t)c * kHW + t] * inv);
}

// ---- normalize refs over C, write bf16 rows R[n*HW+hw][c] row-major ----
__global__ void norm_refs_kernel(const float* __restrict__ refs,
                                 __bf16* __restrict__ Rm) {
  int t = blockIdx.x * blockDim.x + threadIdx.x;   // 0..131071
  int n = t >> 12, hw = t & (kHW - 1);
  const float* base = refs + (size_t)n * kC * kHW + hw;
  float ss = 0.f;
  for (int c = 0; c < kC; ++c) {
    float x = base[(size_t)c * kHW];
    ss += x * x;
  }
  float inv = 1.f / fmaxf(sqrtf(ss), kEps);
  for (int c = 0; c < kC; ++c)
    Rm[(size_t)t * kC + c] = (__bf16)(base[(size_t)c * kHW] * inv);
}

__global__ void init_out_kernel(float* __restrict__ out) {
  int t = blockIdx.x * blockDim.x + threadIdx.x;
  out[t] = -__builtin_inff();
}

// ---- bf16 WMMA GEMM + running max ----
// M-blocked x2 (B register reuse). A sched_barrier(0) between the B-load
// phase and the WMMA phase forces all 16 global_load_b128 of an iteration
// to issue first (8 live fragments, in-order completion), so the WMMAs can
// use staggered partial s_wait_loadcnt instead of full drains.
// block = 8 waves: wave = (qpair 0..3, rcol 0..1); each wave owns 32 queries.
// grid  = (refs/kRB, queries/kQB) = (64, 32)
__global__ __launch_bounds__(256) void sim_max_bf16_wmma(
    const __bf16* __restrict__ Q, const __bf16* __restrict__ R,
    float* __restrict__ out) {
  const int lane   = threadIdx.x & 31;
  const int wave   = threadIdx.x >> 5;
  const int qpair  = wave & 3;
  const int rcol   = wave >> 2;
  const int lane16 = lane & 15;
  const int hi     = lane >> 4;        // 0: lanes 0-15, 1: lanes 16-31

  const int q0 = blockIdx.y * kQB + qpair * 32;   // first of 32 queries

  // Two A tiles (2 x 16 queries x 256 K) resident in registers.
  // ISA 16-bit A layout: lanes 0-15 hold K0-7 / K16-23, lanes 16-31 K8-15 / K24-31.
  v16bf a0[kKC], a1[kKC];
  {
    const char* qrow0 = (const char*)(Q + (size_t)(q0 + lane16) * kC);
    const char* qrow1 = qrow0 + (size_t)16 * kC * sizeof(__bf16);
    #pragma unroll
    for (int kc = 0; kc < kKC; ++kc) {
      FragU f0, f1;
      f0.u[0] = *(const uint4*)(qrow0 + kc * 64 + hi * 16);
      f0.u[1] = *(const uint4*)(qrow0 + kc * 64 + hi * 16 + 32);
      f1.u[0] = *(const uint4*)(qrow1 + kc * 64 + hi * 16);
      f1.u[1] = *(const uint4*)(qrow1 + kc * 64 + hi * 16 + 32);
      a0[kc] = f0.v;
      a1[kc] = f1.v;
    }
  }

  float rmax0[8], rmax1[8];
  #pragma unroll
  for (int r = 0; r < 8; ++r) {
    rmax0[r] = -__builtin_inff();
    rmax1[r] = -__builtin_inff();
  }

  // ISA 16-bit B layout: lane n = column n; lanes 0-15 hold K0-7 / K16-23, lanes 16-31 K8-15 / K24-31.
  const char* rptr =
      (const char*)(R + (size_t)(blockIdx.x * kRB + rcol * 16 + lane16) * kC) +
      hi * 16;

  for (int it = 0; it < kRB / 32; ++it) {
    // Phase 1: issue the whole iteration's B tile as one deep load clause,
    // plus an L0 warm-up prefetch for the next iteration's slab (16 KB ahead).
    FragU b[kKC];
    #pragma unroll
    for (int kc = 0; kc < kKC; ++kc) {
      b[kc].u[0] = *(const uint4*)(rptr + kc * 64);
      b[kc].u[1] = *(const uint4*)(rptr + kc * 64 + 32);
    }
    __builtin_prefetch(rptr + (size_t)32 * kC * sizeof(__bf16), 0, 3);

    // Hard scheduling fence: no WMMA may be hoisted above, no load sunk below.
    __builtin_amdgcn_sched_barrier(0);

    // Phase 2: 16 WMMAs; each fragment feeds two independent accumulators,
    // overlapping with the later fragments' loads still in flight.
    v8f acc0 = {};
    v8f acc1 = {};
    #pragma unroll
    for (int kc = 0; kc < kKC; ++kc) {
      acc0 = __builtin_amdgcn_wmma_f32_16x16x32_bf16(
          false, a0[kc], false, b[kc].v, (short)0, acc0, false, false);
      acc1 = __builtin_amdgcn_wmma_f32_16x16x32_bf16(
          false, a1[kc], false, b[kc].v, (short)0, acc1, false, false);
    }

    // fold this tile's 16 columns into per-lane running maxima;
    // cross-lane reduction deferred to the end (keeps inner loop WMMA-bound)
    #pragma unroll
    for (int r = 0; r < 8; ++r) {
      rmax0[r] = fmaxf(rmax0[r], acc0[r]);
      rmax1[r] = fmaxf(rmax1[r], acc1[r]);
    }
    rptr += (size_t)32 * kC * sizeof(__bf16);   // advance 32 reference rows
  }

  // cross-lane max within each 16-lane half (the 16 columns of each C tile);
  // xor masks 1,2,4,8 stay inside each 16-lane group
  #pragma unroll
  for (int r = 0; r < 8; ++r) {
    float v0 = rmax0[r];
    v0 = fmaxf(v0, __shfl_xor(v0, 1));
    v0 = fmaxf(v0, __shfl_xor(v0, 2));
    v0 = fmaxf(v0, __shfl_xor(v0, 4));
    v0 = fmaxf(v0, __shfl_xor(v0, 8));
    rmax0[r] = v0;
    float v1 = rmax1[r];
    v1 = fmaxf(v1, __shfl_xor(v1, 1));
    v1 = fmaxf(v1, __shfl_xor(v1, 2));
    v1 = fmaxf(v1, __shfl_xor(v1, 4));
    v1 = fmaxf(v1, __shfl_xor(v1, 8));
    rmax1[r] = v1;
  }

  // C layout: VGPR r holds row M=r (lanes 0-15) and M=r+8 (lanes 16-31)
  if (lane16 == 0) {
    float* dst0 = out + q0 + hi * 8;
    float* dst1 = out + q0 + 16 + hi * 8;
    #pragma unroll
    for (int r = 0; r < 8; ++r) {
      __hip_atomic_fetch_max(dst0 + r, rmax0[r], __ATOMIC_RELAXED,
                             __HIP_MEMORY_SCOPE_AGENT);
      __hip_atomic_fetch_max(dst1 + r, rmax1[r], __ATOMIC_RELAXED,
                             __HIP_MEMORY_SCOPE_AGENT);
    }
  }
}

extern "C" void kernel_launch(void* const* d_in, const int* in_sizes, int n_in,
                              void* d_out, int out_size, void* d_ws, size_t ws_size,
                              hipStream_t stream) {
  (void)in_sizes; (void)n_in; (void)out_size; (void)ws_size;
  const float* refs = (const float*)d_in[0];  // (N, C, H, W) f32
  const float* img  = (const float*)d_in[1];  // (C, H, W) f32
  float* out = (float*)d_out;                 // (H, W) f32

  char* ws = (char*)d_ws;
  __bf16* Q = (__bf16*)ws;                                 // 4096*256*2  = 2 MB
  __bf16* R = (__bf16*)(ws + (size_t)kHW * kC * 2);        // 131072*256*2 = 64 MB

  hipLaunchKernelGGL(norm_img_kernel,  dim3(kHW / 256), dim3(256), 0, stream, img, Q);
  hipLaunchKernelGGL(norm_refs_kernel, dim3(kR / 256),  dim3(256), 0, stream, refs, R);
  hipLaunchKernelGGL(init_out_kernel,  dim3(kHW / 256), dim3(256), 0, stream, out);

  dim3 grid(kR / kRB, kHW / kQB);   // (64, 32)
  hipLaunchKernelGGL(sim_max_bf16_wmma, grid, dim3(256), 0, stream, Q, R, out);
}
